// CausalGAT_29850022707669
// MI455X (gfx1250) — matrix operand order repaired and checked by reading, not compiled, not run on hardware
//
#include <hip/hip_runtime.h>
#include <hip/hip_bf16.h>
#include <stdint.h>

typedef __attribute__((ext_vector_type(16))) _Float16 v16h;
typedef __attribute__((ext_vector_type(8)))  _Float16 v8h;
typedef __attribute__((ext_vector_type(8)))  float    v8f;

#define NEG_SLOPE 0.2f

__device__ __forceinline__ unsigned enc_f(float f) {
  unsigned u = __float_as_uint(f);
  return (u & 0x80000000u) ? ~u : (u | 0x80000000u);
}
__device__ __forceinline__ float dec_f(unsigned e) {
  return __uint_as_float((e & 0x80000000u) ? (e & 0x7FFFFFFFu) : ~e);
}
__device__ __forceinline__ float elu_f(float x) { return x > 0.f ? x : (__expf(x) - 1.f); }

__device__ __forceinline__ v16h cat8(v8h a, v8h b) {
  return __builtin_shufflevector(a, b, 0, 1, 2, 3, 4, 5, 6, 7, 8, 9, 10, 11, 12, 13, 14, 15);
}

// ---------------------------------------------------------------------------
// GEMM1: C[N,256] = A[N,128] * B[128,256]; f16 WMMA, f32 accumulate.
// grid.x = N/16 M-tiles; blockDim = 512 (16 waves; wave w owns N-tile w).
// A tile (16x128) staged once per block; B staged per-32-row K slab,
// transposed + padded in LDS so fragments are contiguous ds_load_b128 reads.
// ---------------------------------------------------------------------------
__global__ __launch_bounds__(512) void gemm_feat_kernel(const float* __restrict__ A,
                                                        const float* __restrict__ B,
                                                        float* __restrict__ C, int N) {
  __shared__ __align__(16) _Float16 As[16 * 128];    // row-major 16x128 (4 KB)
  __shared__ __align__(16) _Float16 BsT[256][40];    // B^T slab: [col][k], pad->40 (20 KB)
  const int tid  = threadIdx.x;
  const int wav  = tid >> 5;
  const int lane = tid & 31;
  const int m0 = blockIdx.x * 16, n0 = wav * 16;
  const int row = lane & 15;
  const int hi  = lane >> 4;      // 0 or 1
  const int ka  = hi * 8;         // A: elems 0..7 -> K=ka.., elems 8..15 -> K=ka+16..
  const int kb  = hi * 8;         // B: elems 0..7 -> K=kb.., elems 8..15 -> K=kb+16.., col = lane&15
  // stage A tile (2048 elements / 512 threads)
  for (int i = tid; i < 16 * 128; i += 512) {
    int r = i >> 7, k = i & 127;
    As[i] = (_Float16)A[(size_t)(m0 + r) * 128 + k];
  }
  v8f acc = {};
  for (int ks = 0; ks < 128; ks += 32) {
    __syncthreads();               // Bs reusable / As visible
    for (int i = tid; i < 32 * 256; i += 512) {   // coalesced global read
      int r = i >> 8, c = i & 255;
      BsT[c][r] = (_Float16)B[(size_t)(ks + r) * 256 + c];
    }
    __syncthreads();
    const int arow = row * 128 + ks + ka;
    v16h af = cat8(*(const v8h*)&As[arow], *(const v8h*)&As[arow + 16]);
    const int col = n0 + row;
    v16h bf = cat8(*(const v8h*)&BsT[col][kb], *(const v8h*)&BsT[col][kb + 16]);
    acc = __builtin_amdgcn_wmma_f32_16x16x32_f16(false, af, false, bf, (short)0, acc,
                                                 false, false);
  }
#pragma unroll
  for (int i = 0; i < 8; ++i)     // D: vgpr i -> M = hi*8 + i, col = lane&15
    C[(size_t)(m0 + hi * 8 + i) * 256 + (n0 + row)] = acc[i];
}

// ---------------------------------------------------------------------------
// GEMM2: out[N,16] = (ZA (+ZB))[N,256] * W[256,16]; one wave per 16x16 tile.
// W^T staged once per block in LDS (8.25 KB), contiguous fragment reads.
// ---------------------------------------------------------------------------
__global__ __launch_bounds__(256) void gemm_small_kernel(const float* __restrict__ ZA,
                                                         const float* __restrict__ ZB,
                                                         int addB,
                                                         const float* __restrict__ W,
                                                         float* __restrict__ out, int N) {
  __shared__ __align__(16) _Float16 WsT[16][264];   // [col][k], pad 256->264
  const int tid = threadIdx.x;
  for (int i = tid; i < 256 * 16; i += 256) {       // coalesced: c fastest
    int k = i >> 4, c = i & 15;
    WsT[c][k] = (_Float16)W[(size_t)k * 16 + c];
  }
  __syncthreads();
  const int mt = (int)((blockIdx.x * blockDim.x + tid) >> 5);  // wave-uniform
  if (mt >= (N >> 4)) return;
  const int lane = tid & 31;
  const int m0 = mt * 16;
  const int row = lane & 15;
  const int hi  = lane >> 4;
  const int ka = hi * 8, kb = hi * 8;
  v8f acc = {};
  for (int ks = 0; ks < 256; ks += 32) {
    v16h af, bf;
    const size_t abase = (size_t)(m0 + row) * 256 + ks + ka;
#pragma unroll
    for (int j = 0; j < 8; ++j) {
      float v0 = ZA[abase + j];
      float v1 = ZA[abase + j + 16];
      if (addB) { v0 += ZB[abase + j]; v1 += ZB[abase + j + 16]; }
      af[j] = (_Float16)v0; af[j + 8] = (_Float16)v1;
    }
    bf = cat8(*(const v8h*)&WsT[row][ks + kb], *(const v8h*)&WsT[row][ks + kb + 16]);
    acc = __builtin_amdgcn_wmma_f32_16x16x32_f16(false, af, false, bf, (short)0, acc,
                                                 false, false);
  }
#pragma unroll
  for (int i = 0; i < 8; ++i)
    out[(size_t)(m0 + hi * 8 + i) * 16 + row] = acc[i];
}

// ---------------------------------------------------------------------------
// Per-node attention logits: al_s[n,h] = <hlin[n,h,:], a_src[h,:]>, same dst.
// ---------------------------------------------------------------------------
__global__ void al_kernel(const float* __restrict__ hlin, const float* __restrict__ asr,
                          const float* __restrict__ ads, float* __restrict__ als,
                          float* __restrict__ ald, int N, int H, int C) {
  int t = blockIdx.x * blockDim.x + threadIdx.x;
  if (t >= N * H) return;
  int n = t / H, h = t - n * H;
  const float* hp = hlin + ((size_t)n * H + h) * C;
  float s = 0.f, d = 0.f;
  for (int c = 0; c < C; ++c) {
    s = fmaf(hp[c], asr[h * C + c], s);
    d = fmaf(hp[c], ads[h * C + c], d);
  }
  als[t] = s; ald[t] = d;
}

// Pass 1: e = leaky_relu(al_s[src]+al_d[dst]); store e; segment-max over dst.
__global__ void edge_score_kernel(const long long* __restrict__ ei, int E, int EE,
                                  const float* __restrict__ als, const float* __restrict__ ald,
                                  int H, float* __restrict__ estore,
                                  unsigned* __restrict__ menc) {
  int t = blockIdx.x * blockDim.x + threadIdx.x;
  if (t >= EE * H) return;
  int e = t / H, h = t - e * H;
  int s, d;
  if (e < E) { s = (int)ei[e]; d = (int)ei[E + e]; } else { s = d = e - E; }
  float v = als[(size_t)s * H + h] + ald[(size_t)d * H + h];
  v = v > 0.f ? v : NEG_SLOPE * v;
  estore[t] = v;
  atomicMax(&menc[(size_t)d * H + h], enc_f(v));
}

// Pass 2: p = exp(e - m[dst]); store p; segment-sum over dst.
__global__ void edge_norm_kernel(const long long* __restrict__ ei, int E, int EE, int H,
                                 float* __restrict__ estore,
                                 const unsigned* __restrict__ menc,
                                 float* __restrict__ ssum) {
  int t = blockIdx.x * blockDim.x + threadIdx.x;
  if (t >= EE * H) return;
  int e = t / H, h = t - e * H;
  int d = (e < E) ? (int)ei[E + e] : (e - E);
  float m = dec_f(menc[(size_t)d * H + h]);
  float p = __expf(estore[t] - m);
  estore[t] = p;
  atomicAdd(&ssum[(size_t)d * H + h], p);
}

// Pass 3: agg[dst,h,c] += alpha * hlin[src,h,c].  One thread per (e,h,c):
// a wave's 32 loads and 32 atomic f32 adds hit consecutive addresses.
__global__ void edge_agg_kernel(const long long* __restrict__ ei, int E, int EE, int H, int C,
                                const float* __restrict__ p, const float* __restrict__ ssum,
                                const float* __restrict__ hlin, float* __restrict__ agg) {
  int t = blockIdx.x * blockDim.x + threadIdx.x;
  if (t >= EE * H * C) return;
  int c  = t % C;
  int eh = t / C;
  int e  = eh / H, h = eh - e * H;
  int s, d;
  if (e < E) { s = (int)ei[e]; d = (int)ei[E + e]; } else { s = d = e - E; }
  float alpha = p[eh] / (ssum[(size_t)d * H + h] + 1e-16f);
  atomicAdd(&agg[((size_t)d * H + h) * C + c], hlin[((size_t)s * H + h) * C + c] * alpha);
}

// h = elu(agg1 + b_feat); node_att = softmax(h@Wmlp + bmlp); split xo/xc; elu outs.
__global__ void postL1_kernel(float* __restrict__ aggh, const float* __restrict__ b_feat,
                              const float* __restrict__ Wmlp, const float* __restrict__ bmlp,
                              float* __restrict__ xo, float* __restrict__ xc,
                              float* __restrict__ elu_xo, float* __restrict__ elu_xc, int N) {
  int n = blockIdx.x * blockDim.x + threadIdx.x;
  if (n >= N) return;
  float* hp = aggh + (size_t)n * 256;
  float l0 = bmlp[0], l1 = bmlp[1];
  for (int c = 0; c < 256; ++c) {
    float v = elu_f(hp[c] + b_feat[c]);
    hp[c] = v;
    l0 = fmaf(v, Wmlp[2 * c + 0], l0);
    l1 = fmaf(v, Wmlp[2 * c + 1], l1);
  }
  float mx = fmaxf(l0, l1);
  float e0 = __expf(l0 - mx), e1 = __expf(l1 - mx);
  float inv = 1.f / (e0 + e1);
  float a0 = e0 * inv, a1 = e1 * inv;
  size_t base = (size_t)n * 256;
  for (int c = 0; c < 256; ++c) {
    float v = hp[c];
    float o = a0 * v, q = a1 * v;
    xo[base + c] = o; xc[base + c] = q;
    elu_xo[base + c] = elu_f(o);
    elu_xc[base + c] = elu_f(q);
  }
}

// mean over 2 heads + bias, then log_softmax over 8 channels.
__global__ void finalize2_kernel(const float* __restrict__ agg, const float* __restrict__ b,
                                 float* __restrict__ out, int N) {
  int n = blockIdx.x * blockDim.x + threadIdx.x;
  if (n >= N) return;
  float v[8]; float mx = -1e30f;
  for (int c = 0; c < 8; ++c) {
    v[c] = 0.5f * (agg[(size_t)n * 16 + c] + agg[(size_t)n * 16 + 8 + c]) + b[c];
    mx = fmaxf(mx, v[c]);
  }
  float s = 0.f;
  for (int c = 0; c < 8; ++c) s += __expf(v[c] - mx);
  float ls = __logf(s);
  for (int c = 0; c < 8; ++c) out[(size_t)n * 8 + c] = v[c] - mx - ls;
}

extern "C" void kernel_launch(void* const* d_in, const int* in_sizes, int n_in,
                              void* d_out, int out_size, void* d_ws, size_t ws_size,
                              hipStream_t stream) {
  const float*     x          = (const float*)d_in[0];
  const long long* ei         = (const long long*)d_in[1];   // int64 [2,E]
  const float*     W_feat     = (const float*)d_in[2];
  const float*     a_src_feat = (const float*)d_in[3];
  const float*     a_dst_feat = (const float*)d_in[4];
  const float*     b_feat     = (const float*)d_in[5];
  const float*     W_mlp      = (const float*)d_in[6];
  const float*     b_mlp      = (const float*)d_in[7];
  const float*     W_obj      = (const float*)d_in[8];
  const float*     a_src_obj  = (const float*)d_in[9];
  const float*     a_dst_obj  = (const float*)d_in[10];
  const float*     b_obj      = (const float*)d_in[11];
  const float*     W_ctx      = (const float*)d_in[12];
  const float*     a_src_ctx  = (const float*)d_in[13];
  const float*     a_dst_ctx  = (const float*)d_in[14];
  const float*     b_ctx      = (const float*)d_in[15];
  const float*     W_co       = (const float*)d_in[16];
  const float*     a_src_co   = (const float*)d_in[17];
  const float*     a_dst_co   = (const float*)d_in[18];
  const float*     b_co       = (const float*)d_in[19];

  const int N  = in_sizes[0] / 128;
  const int E  = in_sizes[1] / 2;
  const int EE = E + N;

  // workspace carve-up (floats)
  float*    hlin = (float*)d_ws;                       // [N,256]; reused as xo
  float*    aggh = hlin + (size_t)N * 256;             // agg1, then h (in place)
  float*    xc   = aggh + (size_t)N * 256;             // [N,256]
  float*    p1   = xc + (size_t)N * 256;               // [EE,4]
  float*    als1 = p1 + (size_t)EE * 4;                // [N,4]
  float*    ald1 = als1 + (size_t)N * 4;
  unsigned* m1   = (unsigned*)(ald1 + (size_t)N * 4);  // [N,4]
  float*    s1   = (float*)(m1 + (size_t)N * 4);       // [N,4]
  float*    zlin = s1 + (size_t)N * 4;                 // [N,16]
  float*    als2 = zlin + (size_t)N * 16;              // [N,2]
  float*    ald2 = als2 + (size_t)N * 2;
  unsigned* m2   = (unsigned*)(ald2 + (size_t)N * 2);
  float*    s2   = (float*)(m2 + (size_t)N * 2);
  float*    p2   = s2 + (size_t)N * 2;                 // [EE,2]
  float*    agg2 = p2 + (size_t)EE * 2;                // [N,16]

  float* out       = (float*)d_out;
  float* xo_logis  = out;
  float* xc_logis  = out + (size_t)N * 8;
  float* xco_logis = out + (size_t)2 * N * 8;
  float* elu_xo    = out + (size_t)3 * N * 8;
  float* elu_xc    = elu_xo + (size_t)N * 256;

  // ---- Layer 1: GATConv(128 -> 4x64, concat) ----
  hipMemsetAsync(aggh, 0, (size_t)N * 256 * sizeof(float), stream);
  hipMemsetAsync(m1, 0, (size_t)N * 4 * sizeof(unsigned), stream);
  hipMemsetAsync(s1, 0, (size_t)N * 4 * sizeof(float), stream);

  gemm_feat_kernel<<<N / 16, 512, 0, stream>>>(x, W_feat, hlin, N);
  al_kernel<<<(N * 4 + 255) / 256, 256, 0, stream>>>(hlin, a_src_feat, a_dst_feat,
                                                     als1, ald1, N, 4, 64);
  {
    int tot = EE * 4, blk = (tot + 255) / 256;
    edge_score_kernel<<<blk, 256, 0, stream>>>(ei, E, EE, als1, ald1, 4, p1, m1);
    edge_norm_kernel<<<blk, 256, 0, stream>>>(ei, E, EE, 4, p1, m1, s1);
    int tot3 = EE * 4 * 64, blk3 = (tot3 + 255) / 256;
    edge_agg_kernel<<<blk3, 256, 0, stream>>>(ei, E, EE, 4, 64, p1, s1, hlin, aggh);
  }
  // h + node-att split (xo overwrites hlin: safe, stream-ordered after edge_agg)
  postL1_kernel<<<(N + 127) / 128, 128, 0, stream>>>(aggh, b_feat, W_mlp, b_mlp,
                                                     hlin, xc, elu_xo, elu_xc, N);

  // ---- Layer 2: three GATConv(256 -> mean of 2x8) ----
  auto run_conv2 = [&](const float* za, const float* zb, int addB, const float* W,
                       const float* asr, const float* ads, const float* b, float* lo) {
    hipMemsetAsync(agg2, 0, (size_t)N * 16 * sizeof(float), stream);
    hipMemsetAsync(m2, 0, (size_t)N * 2 * sizeof(unsigned), stream);
    hipMemsetAsync(s2, 0, (size_t)N * 2 * sizeof(float), stream);
    int waves = N / 16, thr = waves * 32;
    gemm_small_kernel<<<(thr + 255) / 256, 256, 0, stream>>>(za, zb, addB, W, zlin, N);
    al_kernel<<<(N * 2 + 255) / 256, 256, 0, stream>>>(zlin, asr, ads, als2, ald2, N, 2, 8);
    int tot = EE * 2, blk = (tot + 255) / 256;
    edge_score_kernel<<<blk, 256, 0, stream>>>(ei, E, EE, als2, ald2, 2, p2, m2);
    edge_norm_kernel<<<blk, 256, 0, stream>>>(ei, E, EE, 2, p2, m2, s2);
    int tot3 = EE * 2 * 8, blk3 = (tot3 + 255) / 256;
    edge_agg_kernel<<<blk3, 256, 0, stream>>>(ei, E, EE, 2, 8, p2, s2, zlin, agg2);
    finalize2_kernel<<<(N + 127) / 128, 128, 0, stream>>>(agg2, b, lo, N);
  };

  run_conv2(hlin, nullptr, 0, W_obj, a_src_obj, a_dst_obj, b_obj, xo_logis);  // xo
  run_conv2(xc,   nullptr, 0, W_ctx, a_src_ctx, a_dst_ctx, b_ctx, xc_logis);  // xc
  run_conv2(hlin, xc,      1, W_co,  a_src_co,  a_dst_co,  b_co,  xco_logis); // xo+xc
}
